// WindowAttention_81819126988802
// MI455X (gfx1250) — matrix operand order repaired and checked by reading, not compiled, not run on hardware
//
#include <hip/hip_runtime.h>
#include <hip/hip_bf16.h>

// ---------------------------------------------------------------------------
// SwinV2 window attention for MI455X (gfx1250, wave32, WMMA).
// All GEMMs use v_wmma_f32_16x16x32_f16 (f16 in, f32 accumulate).
// Attention stages K and V tiles into LDS with GLOBAL_LOAD_ASYNC_TO_LDS_B128
// (ASYNCcnt) when the builtin is available.
// ---------------------------------------------------------------------------

typedef _Float16 half_t;
typedef __attribute__((ext_vector_type(16))) _Float16 v16h;
typedef __attribute__((ext_vector_type(8)))  float    v8f;
typedef __attribute__((ext_vector_type(4)))  int      v4i;

#define WS_SIDE 12
#define NTOK    144
#define CCH     384
#define NH      12
#define HD      32
#define NWIN    512
#define ROWS    (NWIN * NTOK)       // 73728
#define QKVC    (3 * CCH)           // 1152

#if defined(__has_builtin)
#  if __has_builtin(__builtin_amdgcn_global_load_async_to_lds_b128) && \
      __has_builtin(__builtin_amdgcn_s_wait_asynccnt)
#    define ATHENA_ASYNC_LDS 1
#  endif
#endif

__device__ __forceinline__ v8f wmma_f16(v16h a, v16h b, v8f c) {
  return __builtin_amdgcn_wmma_f32_16x16x32_f16(
      /*neg_a=*/false, a, /*neg_b=*/false, b,
      /*c_mod=*/(short)0, c, /*reuse_a=*/false, /*reuse_b=*/false);
}

// A fragment (16x32, f16 source, row-major, stride ld in elements).
// lane m = L%16; K = 16*(L>=16) + e -> 16 contiguous halves per lane.
__device__ __forceinline__ v16h loadA_f16(const half_t* __restrict__ p, int ld) {
  const int lane = threadIdx.x & 31;
  const half_t* r = p + (lane & 15) * ld;
  const int koff = (lane & 16) ? 16 : 0;
  v16h a;
#pragma unroll
  for (int e = 0; e < 16; ++e) a[e] = r[koff + e];
  return a;
}

// A fragment where only K in [0,16) is valid (tail); K>=16 elements are zero.
__device__ __forceinline__ v16h loadA_f16_tail(const half_t* __restrict__ p, int ld) {
  const int lane = threadIdx.x & 31;
  const half_t* r = p + (lane & 15) * ld;
  const bool hiHalf = (lane & 16) != 0;
  v16h a;
#pragma unroll
  for (int e = 0; e < 16; ++e) a[e] = hiHalf ? (half_t)0.f : r[e];   // K = e in [0,16); K >= 16 -> 0
  return a;
}

// A fragment from f32 source (converted on the fly).
__device__ __forceinline__ v16h loadA_f32(const float* __restrict__ p, int ld) {
  const int lane = threadIdx.x & 31;
  const float* r = p + (lane & 15) * ld;
  const int koff = (lane & 16) ? 16 : 0;
  v16h a;
#pragma unroll
  for (int e = 0; e < 16; ++e) a[e] = (half_t)r[koff + e];
  return a;
}

// B fragment (32x16) where source is row-major [N][K] (i.e. B = srcT):
// lane n = L%16 selects source row; K = 16*(L>=16) + e -> 16 contiguous halves.
__device__ __forceinline__ v16h loadB_nk(const half_t* __restrict__ p, int ld) {
  const int lane = threadIdx.x & 31;
  const half_t* r = p + (lane & 15) * ld + ((lane & 16) ? 16 : 0);
  v16h b;
#pragma unroll
  for (int e = 0; e < 16; ++e) b[e] = r[e];
  return b;
}

// B fragment (32x16) where source is row-major [K][N].
__device__ __forceinline__ v16h loadB_kn(const half_t* __restrict__ p, int ld) {
  const int lane = threadIdx.x & 31;
  const int n = lane & 15;
  const int kb = (lane & 16) ? 16 : 0;
  v16h b;
#pragma unroll
  for (int e = 0; e < 16; ++e) b[e] = p[(kb + e) * ld + n];
  return b;
}

// B fragment [K][N] where only K in [0,16) is valid; K>=16 -> zero.
// Lanes >= 16 (kb=16) still issue in-range loads, then select 0: no divergence.
__device__ __forceinline__ v16h loadB_kn_tail(const half_t* __restrict__ p, int ld) {
  const int lane = threadIdx.x & 31;
  const int n = lane & 15;
  const bool hiHalf = (lane & 16) != 0;
  v16h b;
#pragma unroll
  for (int e = 0; e < 16; ++e) {
    half_t v = p[e * ld + n];
    b[e] = hiHalf ? (half_t)0.f : v;
  }
  return b;
}

// ---------------------------------------------------------------------------
// Kernel 0: convert weights to f16 (stay L2-resident thereafter).
// ---------------------------------------------------------------------------
__global__ __launch_bounds__(256) void k_wcvt(const float* __restrict__ qkvw,
                                              const float* __restrict__ projw,
                                              half_t* __restrict__ wq,
                                              half_t* __restrict__ wp) {
  const int i = blockIdx.x * 256 + threadIdx.x;
  if (i < QKVC * CCH) wq[i] = (half_t)qkvw[i];
  if (i < CCH * CCH)  wp[i] = (half_t)projw[i];
}

// ---------------------------------------------------------------------------
// Kernel 1: CPB MLP table: tab[t][h], t in [0,529), h in [0,12).
// ---------------------------------------------------------------------------
__global__ __launch_bounds__(256) void k_cpb(const float* __restrict__ w1,
                                             const float* __restrict__ b1,
                                             const float* __restrict__ w2,
                                             float* __restrict__ tab) {
  const int i = blockIdx.x * 256 + threadIdx.x;
  if (i >= 529 * NH) return;
  const int t = i / NH, h = i % NH;
  const float c0 = ((float)(t / 23) - 11.f) * (12.f / 11.f);
  const float c1 = ((float)(t % 23) - 11.f) * (12.f / 11.f);
  const float t0 = copysignf(log2f(fabsf(c0) + 1.f) * (1.f / 3.f), c0);
  const float t1 = copysignf(log2f(fabsf(c1) + 1.f) * (1.f / 3.f), c1);
  float acc = 0.f;
  for (int k = 0; k < 512; ++k) {
    float hid = fmaxf(t0 * w1[2 * k] + t1 * w1[2 * k + 1] + b1[k], 0.f);
    acc = fmaf(hid, w2[h * 512 + k], acc);
  }
  tab[t * NH + h] = acc;
}

// ---------------------------------------------------------------------------
// Kernel 2: combined additive term cmb[w][h][n][m] = 16*sigmoid(bias)+mask.
// 4 MB -> lives in L2 while attention re-reads it.
// ---------------------------------------------------------------------------
__global__ __launch_bounds__(256) void k_combine(const float* __restrict__ tab,
                                                 const float* __restrict__ mask,
                                                 float* __restrict__ cmb) {
  const int i = blockIdx.x * 256 + threadIdx.x;
  if (i >= 4 * NH * NTOK * NTOK) return;
  const int m = i % NTOK;
  const int n = (i / NTOK) % NTOK;
  const int h = (i / (NTOK * NTOK)) % NH;
  const int w = i / (NTOK * NTOK * NH);
  const int dy = n / WS_SIDE - m / WS_SIDE + (WS_SIDE - 1);
  const int dx = n % WS_SIDE - m % WS_SIDE + (WS_SIDE - 1);
  const float b = tab[(dy * 23 + dx) * NH + h];
  cmb[i] = 16.f / (1.f + __expf(-b)) + mask[(w * NTOK + n) * NTOK + m];
}

// ---------------------------------------------------------------------------
// Kernel 3: QKV GEMM (73728x384 @ 384x1152) + bias, fused L2-norm for q,k.
// Block = 4 waves, tile 64 rows x 64 cols (exactly 2 heads of one of q/k/v).
// Writes f16 per-head layout [bw][h][n][d].
// ---------------------------------------------------------------------------
__global__ __launch_bounds__(128) void k_qkv(const float* __restrict__ x,
                                             const half_t* __restrict__ wqkv,
                                             const float* __restrict__ qkv_b,
                                             half_t* __restrict__ qg,
                                             half_t* __restrict__ kg,
                                             half_t* __restrict__ vg) {
  const int rowBase = blockIdx.x * 64;
  const int colBase = blockIdx.y * 64;            // [0,1152)
  const int wv = threadIdx.x >> 5;
  const int lane = threadIdx.x & 31;
  const int row0 = rowBase + wv * 16;
  const int ncol = lane & 15;
  const int hi = (lane & 16) ? 8 : 0;

  v8f acc[4];
#pragma unroll
  for (int ct = 0; ct < 4; ++ct) acc[ct] = (v8f){};

  const float* xrow = x + (size_t)(row0 + (lane & 15)) * CCH;

#pragma unroll 4
  for (int kt = 0; kt < CCH / 32; ++kt) {
    if (kt + 1 < CCH / 32)                        // hide next A-tile latency
      __builtin_prefetch((const void*)(xrow + (kt + 1) * 32), 0, 0);
    v16h a = loadA_f32(x + (size_t)row0 * CCH + kt * 32, CCH);
#pragma unroll
    for (int ct = 0; ct < 4; ++ct) {
      v16h b = loadB_nk(wqkv + (size_t)(colBase + ct * 16) * CCH + kt * 32, CCH);
      acc[ct] = wmma_f16(a, b, acc[ct]);
    }
  }

  // + qkv bias (per output column)
#pragma unroll
  for (int ct = 0; ct < 4; ++ct) {
    const float bias = qkv_b[colBase + ct * 16 + ncol];
#pragma unroll
    for (int r = 0; r < 8; ++r) acc[ct][r] += bias;
  }

  const int part = colBase / CCH;                 // 0=q 1=k 2=v
  const int pc = colBase - part * CCH;            // head-pair offset, mult of 64

  if (part < 2) {                                  // cosine-attn L2 normalize
#pragma unroll
    for (int hp = 0; hp < 2; ++hp) {
#pragma unroll
      for (int r = 0; r < 8; ++r) {
        float ss = acc[2 * hp][r] * acc[2 * hp][r] +
                   acc[2 * hp + 1][r] * acc[2 * hp + 1][r];
        ss += __shfl_xor(ss, 1, 32);
        ss += __shfl_xor(ss, 2, 32);
        ss += __shfl_xor(ss, 4, 32);
        ss += __shfl_xor(ss, 8, 32);               // stays within 16-lane half
        const float inv = 1.f / fmaxf(sqrtf(ss), 1e-12f);
        acc[2 * hp][r] *= inv;
        acc[2 * hp + 1][r] *= inv;
      }
    }
  }

  half_t* dst = (part == 0) ? qg : (part == 1) ? kg : vg;
#pragma unroll
  for (int ct = 0; ct < 4; ++ct) {
    const int h = (pc + ct * 16) / HD;
    const int d = (pc + ct * 16) % HD + ncol;
#pragma unroll
    for (int r = 0; r < 8; ++r) {
      const int g = row0 + r + hi;
      const int bw = g / NTOK, n = g % NTOK;
      dst[(((size_t)bw * NH + h) * NTOK + n) * HD + d] = (half_t)acc[ct][r];
    }
  }
}

// ---------------------------------------------------------------------------
// Kernel 4: fused attention, one block per (window, head). 9 waves; wave w
// owns rows [16w,16w+16). K and V head tiles (9216 B each) are staged in LDS
// via GLOBAL_LOAD_ASYNC_TO_LDS_B128 (ASYNCcnt) overlapping the q-fragment
// global loads; S = q@k^T from LDS, epilogue scale+bias+mask, in-register
// softmax (shfl within 16-lane halves), P f16 in LDS, then P@V with 4 full
// K=32 steps + one zero-masked K=16 tail (no padding needed).
// LDS: 144*144*2 + 2*144*32*2 = 59904 B.
// ---------------------------------------------------------------------------
__global__ __launch_bounds__(288) void k_attn(const half_t* __restrict__ qg,
                                              const half_t* __restrict__ kg,
                                              const half_t* __restrict__ vg,
                                              const float* __restrict__ cmb,
                                              const float* __restrict__ ls,
                                              half_t* __restrict__ aout) {
  __shared__ half_t Pl[NTOK * NTOK];
  __shared__ half_t Kl[NTOK * HD];
  __shared__ half_t Vl[NTOK * HD];

  const int bw = blockIdx.x / NH;
  const int h  = blockIdx.x % NH;
  const int wm = bw & 3;                           // window index in mask group
  const size_t hb = (((size_t)bw * NH + h)) * (NTOK * HD);

  // stage K and V head tiles into LDS (9216 B each = 576 b128; 2 per thread)
#ifdef ATHENA_ASYNC_LDS
  {
    const char* gk = (const char*)(kg + hb);
    const char* gv = (const char*)(vg + hb);
    char* lk = (char*)Kl;
    char* lv = (char*)Vl;
#pragma unroll
    for (int it = 0; it < 2; ++it) {
      const int off = ((int)threadIdx.x + it * 288) * 16;
      __builtin_amdgcn_global_load_async_to_lds_b128(
          (__attribute__((address_space(1))) v4i*)(gk + off),
          (__attribute__((address_space(3))) v4i*)(lk + off), 0, 0);
      __builtin_amdgcn_global_load_async_to_lds_b128(
          (__attribute__((address_space(1))) v4i*)(gv + off),
          (__attribute__((address_space(3))) v4i*)(lv + off), 0, 0);
    }
  }
#else
  for (int i = threadIdx.x; i < NTOK * HD; i += 288) {
    Kl[i] = kg[hb + i];
    Vl[i] = vg[hb + i];
  }
#endif

  const float scale = __expf(fminf(ls[h], 4.6051702f));   // log(1/0.01)
  const int wv = threadIdx.x >> 5;
  const int lane = threadIdx.x & 31;
  const int m0 = wv * 16;
  const int ncol = lane & 15;
  const int hi = (lane & 16) ? 8 : 0;
  const float* cb = cmb + ((size_t)(wm * NH + h)) * (NTOK * NTOK);

  // q fragment from global overlaps with the async K/V staging
  v16h aq = loadA_f16(qg + hb + (size_t)m0 * HD, HD);

#ifdef ATHENA_ASYNC_LDS
  __builtin_amdgcn_s_wait_asynccnt(0);             // this wave's asyncs landed
#endif
  __syncthreads();                                 // all waves' staging visible

  float s[9][8];
#pragma unroll
  for (int ct = 0; ct < 9; ++ct) {
    v16h bk = loadB_nk(Kl + ct * 16 * HD, HD);
    v8f c = (v8f){};
    c = wmma_f16(aq, bk, c);
#pragma unroll
    for (int r = 0; r < 8; ++r) {
      const int row = m0 + r + hi;
      s[ct][r] = c[r] * scale + cb[(size_t)row * NTOK + ct * 16 + ncol];
    }
  }

  // row softmax + write P (f16)
#pragma unroll
  for (int r = 0; r < 8; ++r) {
    float mx = -3.0e38f;
#pragma unroll
    for (int ct = 0; ct < 9; ++ct) mx = fmaxf(mx, s[ct][r]);
    mx = fmaxf(mx, __shfl_xor(mx, 1, 32));
    mx = fmaxf(mx, __shfl_xor(mx, 2, 32));
    mx = fmaxf(mx, __shfl_xor(mx, 4, 32));
    mx = fmaxf(mx, __shfl_xor(mx, 8, 32));
    float sum = 0.f;
#pragma unroll
    for (int ct = 0; ct < 9; ++ct) {
      const float e = __expf(s[ct][r] - mx);
      s[ct][r] = e;
      sum += e;
    }
    sum += __shfl_xor(sum, 1, 32);
    sum += __shfl_xor(sum, 2, 32);
    sum += __shfl_xor(sum, 4, 32);
    sum += __shfl_xor(sum, 8, 32);
    const float inv = 1.f / sum;
    const int row = m0 + r + hi;
#pragma unroll
    for (int ct = 0; ct < 9; ++ct)
      Pl[row * NTOK + ct * 16 + ncol] = (half_t)(s[ct][r] * inv);
  }

  __syncthreads();

  // O = P @ V: 4 full K=32 steps + zero-masked K=16 tail (tokens 128..143)
  v8f o0 = (v8f){}, o1 = (v8f){};
#pragma unroll
  for (int kt = 0; kt < 4; ++kt) {
    v16h ap = loadA_f16(Pl + m0 * NTOK + kt * 32, NTOK);
    v16h b0 = loadB_kn(Vl + kt * 32 * HD, HD);
    v16h b1 = loadB_kn(Vl + kt * 32 * HD + 16, HD);
    o0 = wmma_f16(ap, b0, o0);
    o1 = wmma_f16(ap, b1, o1);
  }
  {
    v16h ap = loadA_f16_tail(Pl + m0 * NTOK + 128, NTOK);
    v16h b0 = loadB_kn_tail(Vl + 128 * HD, HD);
    v16h b1 = loadB_kn_tail(Vl + 128 * HD + 16, HD);
    o0 = wmma_f16(ap, b0, o0);
    o1 = wmma_f16(ap, b1, o1);
  }

  // store f16 attn-out in [bw][n][h*32+d] layout for the proj GEMM
#pragma unroll
  for (int r = 0; r < 8; ++r) {
    const int n = m0 + r + hi;
    const size_t ob = ((size_t)bw * NTOK + n) * CCH + h * HD;
    aout[ob + ncol]      = (half_t)o0[r];
    aout[ob + 16 + ncol] = (half_t)o1[r];
  }
}

// ---------------------------------------------------------------------------
// Kernel 5: projection GEMM (73728x384 @ 384x384) + bias -> f32 output.
// Block = 4 waves, tile 64x64.
// ---------------------------------------------------------------------------
__global__ __launch_bounds__(128) void k_proj(const half_t* __restrict__ a_in,
                                              const half_t* __restrict__ wp,
                                              const float* __restrict__ pb,
                                              float* __restrict__ out) {
  const int rowBase = blockIdx.x * 64;
  const int colBase = blockIdx.y * 64;
  const int wv = threadIdx.x >> 5;
  const int lane = threadIdx.x & 31;
  const int row0 = rowBase + wv * 16;
  const int ncol = lane & 15;
  const int hi = (lane & 16) ? 8 : 0;

  v8f acc[4];
#pragma unroll
  for (int ct = 0; ct < 4; ++ct) acc[ct] = (v8f){};

  const half_t* arow = a_in + (size_t)(row0 + (lane & 15)) * CCH;

#pragma unroll 4
  for (int kt = 0; kt < CCH / 32; ++kt) {
    if (kt + 1 < CCH / 32)
      __builtin_prefetch((const void*)(arow + (kt + 1) * 32), 0, 0);
    v16h a = loadA_f16(a_in + (size_t)row0 * CCH + kt * 32, CCH);
#pragma unroll
    for (int ct = 0; ct < 4; ++ct) {
      v16h b = loadB_nk(wp + (size_t)(colBase + ct * 16) * CCH + kt * 32, CCH);
      acc[ct] = wmma_f16(a, b, acc[ct]);
    }
  }

#pragma unroll
  for (int ct = 0; ct < 4; ++ct) {
    const int col = colBase + ct * 16 + ncol;
    const float bias = pb[col];
#pragma unroll
    for (int r = 0; r < 8; ++r) {
      const int g = row0 + r + hi;
      out[(size_t)g * CCH + col] = acc[ct][r] + bias;
    }
  }
}

// ---------------------------------------------------------------------------
// Host-side launch
// ---------------------------------------------------------------------------
extern "C" void kernel_launch(void* const* d_in, const int* in_sizes, int n_in,
                              void* d_out, int out_size, void* d_ws, size_t ws_size,
                              hipStream_t stream) {
  (void)in_sizes; (void)n_in; (void)out_size; (void)ws_size;

  const float* x     = (const float*)d_in[0];
  const float* mask  = (const float*)d_in[1];
  const float* qkv_w = (const float*)d_in[2];
  const float* qkv_b = (const float*)d_in[3];
  const float* lsc   = (const float*)d_in[4];
  const float* cpbw1 = (const float*)d_in[5];
  const float* cpbb1 = (const float*)d_in[6];
  const float* cpbw2 = (const float*)d_in[7];
  const float* prw   = (const float*)d_in[8];
  const float* prb   = (const float*)d_in[9];
  float* out = (float*)d_out;

  // Workspace layout (all offsets 256B-aligned), total ~232 MB
  char* ws = (char*)d_ws;
  half_t* wq  = (half_t*)(ws);                    //  884,736 B  qkv_w f16
  half_t* wp  = (half_t*)(ws + 884736);           //  294,912 B  proj_w f16
  float*  tab = (float*) (ws + 1179648);          //   25,392 B  CPB table
  float*  cmb = (float*) (ws + 1205248);          // 3,981,312 B bias+mask
  half_t* qg  = (half_t*)(ws + 5186560);          // 56.6 MB    q (normalized)
  half_t* kg  = (half_t*)(ws + 61809664);         // 56.6 MB    k (normalized)
  half_t* vg  = (half_t*)(ws + 118432768);        // 56.6 MB    v
  half_t* ao  = (half_t*)(ws + 175055872);        // 56.6 MB    attn out f16

  k_wcvt   <<<(QKVC * CCH + 255) / 256, 256, 0, stream>>>(qkv_w, prw, wq, wp);
  k_cpb    <<<(529 * NH + 255) / 256,   256, 0, stream>>>(cpbw1, cpbb1, cpbw2, tab);
  k_combine<<<(4 * NH * NTOK * NTOK) / 256, 256, 0, stream>>>(tab, mask, cmb);
  k_qkv    <<<dim3(ROWS / 64, QKVC / 64), 128, 0, stream>>>(x, wq, qkv_b, qg, kg, vg);
  k_attn   <<<NWIN * NH, 288, 0, stream>>>(qg, kg, vg, cmb, lsc, ao);
  k_proj   <<<dim3(ROWS / 64, CCH / 64), 128, 0, stream>>>(ao, wp, prb, out);
}
